// transformerModel_83889301225836
// MI455X (gfx1250) — matrix operand order repaired and checked by reading, not compiled, not run on hardware
//
#include <hip/hip_runtime.h>

// ---------------------------------------------------------------------------
// GPT-2-ish forward (B=2,T=1024,D=768,H=12,DH=64,L=4,FF=3072,V=32000)
// All matmuls via v_wmma_f32_16x16x32_bf16 (wave32 CDNA5), flash attention,
// bf16 data path with fp32 accumulation. GEMM: 32x64 tile/wave, ping-pong
// double-buffered K loop (2 steps/iter, no rotation copies).
// ---------------------------------------------------------------------------

#define BB 2
#define TT 1024
#define DD 768
#define HH 12
#define DHD 64
#define LL 4
#define FFD 3072
#define VV 32000
#define MM (BB*TT)          // 2048 rows
#define QKVN (3*DD)         // 2304

typedef __attribute__((ext_vector_type(8)))  __bf16 bf16x8;
typedef __attribute__((ext_vector_type(16))) __bf16 v16bf;
typedef __attribute__((ext_vector_type(8)))  float  v8f;

static __device__ __forceinline__ __bf16 f2bf(float f) {
    union { float f; unsigned u; } v; v.f = f;
    unsigned r = v.u + 0x7FFFu + ((v.u >> 16) & 1u);   // RNE
    unsigned short h = (unsigned short)(r >> 16);
    return __builtin_bit_cast(__bf16, h);
}
static __device__ __forceinline__ bf16x8 ld8(const __bf16* p)  { return *(const bf16x8*)p; }
static __device__ __forceinline__ v16bf  ld16(const __bf16* p) { return *(const v16bf*)p; }
static __device__ __forceinline__ v16bf  cat16(bf16x8 a, bf16x8 b) {
    return __builtin_shufflevector(a, b, 0,1,2,3,4,5,6,7,8,9,10,11,12,13,14,15);
}
static __device__ __forceinline__ v8f wmma_bf16(v16bf a, v16bf b, v8f c) {
    return __builtin_amdgcn_wmma_f32_16x16x32_bf16(false, a, false, b, (short)0, c, false, false);
}

// ------------------------------- embedding ---------------------------------
__global__ void k_embed(const int* __restrict__ tokens, const float* __restrict__ tok,
                        const float* __restrict__ pos, float* __restrict__ x) {
    size_t idx = (size_t)blockIdx.x * blockDim.x + threadIdx.x;
    if (idx >= (size_t)MM * DD) return;
    int m = (int)(idx / DD), d = (int)(idx % DD);
    int t = m % TT;
    x[idx] = tok[(size_t)tokens[m] * DD + d] + pos[(size_t)t * DD + d];
}

// ------------------------------- layernorm ---------------------------------
__global__ void k_layernorm(const float* __restrict__ x, const float* __restrict__ g,
                            const float* __restrict__ b, __bf16* __restrict__ out) {
    __shared__ float red[256];
    int row = blockIdx.x, tid = threadIdx.x;
    const float* xr = x + (size_t)row * DD;
    float s = 0.f;
    for (int i = tid; i < DD; i += 256) s += xr[i];
    red[tid] = s; __syncthreads();
    for (int st = 128; st > 0; st >>= 1) { if (tid < st) red[tid] += red[tid + st]; __syncthreads(); }
    float mu = red[0] / (float)DD; __syncthreads();
    float vs = 0.f;
    for (int i = tid; i < DD; i += 256) { float d = xr[i] - mu; vs += d * d; }
    red[tid] = vs; __syncthreads();
    for (int st = 128; st > 0; st >>= 1) { if (tid < st) red[tid] += red[tid + st]; __syncthreads(); }
    float rstd = rsqrtf(red[0] / (float)DD + 1e-5f);
    for (int i = tid; i < DD; i += 256)
        out[(size_t)row * DD + i] = f2bf((xr[i] - mu) * rstd * g[i] + b[i]);
}

// -------------------- weight packing into WMMA B-tile layout ----------------
// Packed tile (kt,nt): 32 lanes x 16 contiguous bf16.
//   lane<16 : B[kt*32 + i][nt*16 + lane],      i=0..15
//   lane>=16: B[kt*32 + 16 + i][nt*16+lane-16]
// offset(elems) = ((ntGlobal*nkt + kt)*32 + lane)*16
__global__ void k_pack_b_f32(const float* __restrict__ src, __bf16* __restrict__ dst,
                             int K, int N) {
    int idx = blockIdx.x * blockDim.x + threadIdx.x;
    int nkt = K >> 5, nnt = N >> 4;
    if (idx >= nkt * nnt * 32) return;
    int lane = idx & 31, t = idx >> 5;
    int kt = t % nkt, nt = t / nkt;
    int kbase = kt * 32 + ((lane >> 4) << 4);
    int col = nt * 16 + (lane & 15);
    __bf16* d = dst + (((size_t)nt * nkt + kt) * 32 + lane) * 16;
#pragma unroll
    for (int i = 0; i < 16; ++i) d[i] = f2bf(src[(size_t)(kbase + i) * N + col]);
}

// Wq/Wk/Wv layout [H, K, 64]; logical column n = h*64+e; packed into the
// combined QKV matrix (N=2304) at ntOff (in 16-col tiles).
__global__ void k_pack_b_qkv(const float* __restrict__ src, __bf16* __restrict__ dst,
                             int ntOff) {
    int idx = blockIdx.x * blockDim.x + threadIdx.x;
    const int K = DD, nkt = K >> 5, nntLocal = DD >> 4; // 24, 48
    if (idx >= nkt * nntLocal * 32) return;
    int lane = idx & 31, t = idx >> 5;
    int kt = t % nkt, ntl = t / nkt;
    int kbase = kt * 32 + ((lane >> 4) << 4);
    int ncol = ntl * 16 + (lane & 15);
    int h = ncol >> 6, e = ncol & 63;
    __bf16* d = dst + (((size_t)(ntOff + ntl) * nkt + kt) * 32 + lane) * 16;
#pragma unroll
    for (int i = 0; i < 16; ++i)
        d[i] = f2bf(src[((size_t)h * K + kbase + i) * 64 + e]);
}

// Pack V (bf16, columns 1536.. of qkv activations) per head into B-tiles.
__global__ void k_pack_v(const __bf16* __restrict__ qkv, __bf16* __restrict__ vpack) {
    int idx = blockIdx.x * blockDim.x + threadIdx.x;
    const int nkt = TT >> 5; // 32
    if (idx >= BB * HH * nkt * 4 * 32) return;
    int lane = idx & 31, t = idx >> 5;
    int kt = t & 31; t >>= 5;
    int nt = t & 3;  int bh = t >> 2;
    int b = bh / HH, h = bh % HH;
    int kbase = kt * 32 + ((lane >> 4) << 4);
    int e = nt * 16 + (lane & 15);
    __bf16* d = vpack + (size_t)bh * TT * DHD + (((size_t)nt * nkt + kt) * 32 + lane) * 16;
#pragma unroll
    for (int i = 0; i < 16; ++i)
        d[i] = qkv[((size_t)(b * TT) + kbase + i) * QKVN + 2 * DD + h * 64 + e];
}

// ------------------------------ WMMA GEMM ----------------------------------
// C[M,N] = A[M,K](bf16 row-major) @ Bpacked  (+bias)(+relu)(+residual)
// One wave -> 32x64 tile (2 M-tiles x 4 N-tiles = 8 WMMAs / K-step, B shared
// across M-tiles). K-loop: manual ping-pong double buffering, two K-steps per
// iteration (K/32 is even for all uses), so no register rotation copies and
// loads of step k+1 overlap the WMMAs of step k.
// Block = 8 waves = 256 rows per block.
__global__ void k_gemm(const __bf16* __restrict__ A, const __bf16* __restrict__ Bp,
                       const float* __restrict__ bias, const float* __restrict__ resid,
                       float* __restrict__ outF, __bf16* __restrict__ outB,
                       int N, int K, int relu) {
    int lane = threadIdx.x & 31, w = threadIdx.x >> 5;
    int rowBase = (blockIdx.y * 8 + w) * 32;
    int nt0 = blockIdx.x * 4;
    int nkt = K >> 5;
    int lhalf = lane >> 4, lmod = lane & 15;
    const __bf16* arow0 = A + (size_t)(rowBase + lmod) * K;
    const __bf16* arow1 = A + (size_t)(rowBase + 16 + lmod) * K;
    int aoff0 = lhalf ? 16 : 0, aoff1 = aoff0 + 8;

    auto loadA = [&](int kt, v16bf* a) {
        a[0] = cat16(ld8(arow0 + kt * 32 + aoff0), ld8(arow0 + kt * 32 + aoff1));
        a[1] = cat16(ld8(arow1 + kt * 32 + aoff0), ld8(arow1 + kt * 32 + aoff1));
    };
    auto loadB = [&](int kt, v16bf* b) {
#pragma unroll
        for (int n = 0; n < 4; ++n)
            b[n] = ld16(Bp + (((size_t)(nt0 + n) * nkt + kt) * 32 + lane) * 16);
    };

    v8f acc[2][4] = {};
    auto mma = [&](v16bf* a, v16bf* b) {
#pragma unroll
        for (int n = 0; n < 4; ++n) {
            acc[0][n] = wmma_bf16(a[0], b[n], acc[0][n]);
            acc[1][n] = wmma_bf16(a[1], b[n], acc[1][n]);
        }
    };

    v16bf a0[2], b0[4], a1[2], b1[4];
    loadA(0, a0); loadB(0, b0);
    int kt = 0;
    for (; kt < nkt - 2; kt += 2) {
        loadA(kt + 1, a1); loadB(kt + 1, b1);   // prefetch odd step
        mma(a0, b0);
        loadA(kt + 2, a0); loadB(kt + 2, b0);   // prefetch next even step
        mma(a1, b1);
    }
    loadA(kt + 1, a1); loadB(kt + 1, b1);
    mma(a0, b0);
    mma(a1, b1);

#pragma unroll
    for (int mi = 0; mi < 2; ++mi)
#pragma unroll
        for (int n = 0; n < 4; ++n) {
            int col = (nt0 + n) * 16 + lmod;
            float bs = bias ? bias[col] : 0.f;
#pragma unroll
            for (int r = 0; r < 8; ++r) {
                int row = rowBase + mi * 16 + r + lhalf * 8;
                float v = acc[mi][n][r] + bs;
                if (relu) v = v > 0.f ? v : 0.f;
                if (resid) v += resid[(size_t)row * N + col];
                if (outF) outF[(size_t)row * N + col] = v;
                if (outB) outB[(size_t)row * N + col] = f2bf(v);
            }
        }
}

// --------------------------- flash attention -------------------------------
// One wave = 16 query rows of one (b,h). Online softmax over 32-key chunks.
// Next chunk's K-fragments and this chunk's V-fragments are loaded before the
// softmax VALU work, overlapping VMEM latency with exp/shuffle reductions.
__global__ void k_flash(const __bf16* __restrict__ qkv, const __bf16* __restrict__ vpack,
                        __bf16* __restrict__ oheads) {
    __shared__ __attribute__((aligned(64))) __bf16 lds[8][16 * 32];
    int lane = threadIdx.x & 31, w = threadIdx.x >> 5;
    int qt = blockIdx.x;
    int bh = blockIdx.y * 8 + w;
    int b = bh / HH, h = bh % HH;
    int lhalf = lane >> 4, lmod = lane & 15;
    __bf16* P = lds[w];
    const float scale = 0.125f; // DH^-0.5

    // Q fragments (A-layout), 2 K-chunks over DH=64
    const __bf16* qbase = qkv + ((size_t)(b * TT) + qt * 16 + lmod) * QKVN + h * 64;
    int aoff0 = lhalf ? 16 : 0, aoff1 = aoff0 + 8;
    v16bf aq[2];
#pragma unroll
    for (int c = 0; c < 2; ++c)
        aq[c] = cat16(ld8(qbase + c * 32 + aoff0), ld8(qbase + c * 32 + aoff1));

    auto loadK = [&](int kb, v16bf bk[2][2]) {
#pragma unroll
        for (int nt = 0; nt < 2; ++nt)
#pragma unroll
            for (int c = 0; c < 2; ++c)
                bk[nt][c] = ld16(qkv + ((size_t)(b * TT) + kb + nt * 16 + lmod) * QKVN
                                 + DD + h * 64 + c * 32 + lhalf * 16);
    };

    float mrow[8], lsum[8];
    v8f o[4] = {};
#pragma unroll
    for (int r = 0; r < 8; ++r) { mrow[r] = -INFINITY; lsum[r] = 0.f; }

    const __bf16* vb = vpack + (size_t)bh * TT * DHD;
    int nchunks = ((qt * 16 + 15) >> 5) + 1;

    v16bf bk[2][2];
    loadK(0, bk);
    for (int kc = 0; kc < nchunks; ++kc) {
        int kb = kc * 32;
        v8f s0 = {}, s1 = {};
        s0 = wmma_bf16(aq[0], bk[0][0], s0);
        s0 = wmma_bf16(aq[1], bk[0][1], s0);
        s1 = wmma_bf16(aq[0], bk[1][0], s1);
        s1 = wmma_bf16(aq[1], bk[1][1], s1);

        // prefetch next chunk's K-fragments and this chunk's V-fragments
        v16bf bkn[2][2];
        if (kc + 1 < nchunks) loadK(kb + 32, bkn);
        v16bf bv[4];
#pragma unroll
        for (int nt = 0; nt < 4; ++nt)
            bv[nt] = ld16(vb + (((size_t)nt * (TT >> 5) + kc) * 32 + lane) * 16);

        float p0[8], p1[8], resc[8];
#pragma unroll
        for (int r = 0; r < 8; ++r) {
            int row = qt * 16 + r + lhalf * 8;
            float v0 = s0[r] * scale, v1 = s1[r] * scale;
            if (kb + lmod      > row) v0 = -INFINITY;
            if (kb + 16 + lmod > row) v1 = -INFINITY;
            float mx = fmaxf(v0, v1);
#pragma unroll
            for (int msk = 1; msk <= 8; msk <<= 1) mx = fmaxf(mx, __shfl_xor(mx, msk));
            float mnew = fmaxf(mrow[r], mx);
            resc[r] = __expf(mrow[r] - mnew);
            mrow[r] = mnew;
            p0[r] = __expf(v0 - mnew);
            p1[r] = __expf(v1 - mnew);
            float rs = p0[r] + p1[r];
#pragma unroll
            for (int msk = 1; msk <= 8; msk <<= 1) rs += __shfl_xor(rs, msk);
            lsum[r] = lsum[r] * resc[r] + rs;
        }
#pragma unroll
        for (int n = 0; n < 4; ++n)
#pragma unroll
            for (int r = 0; r < 8; ++r) o[n][r] *= resc[r];

        // P (C-layout) -> LDS -> A-fragment
#pragma unroll
        for (int r = 0; r < 8; ++r) {
            int row = r + lhalf * 8;
            P[row * 32 + lmod]      = f2bf(p0[r]);
            P[row * 32 + 16 + lmod] = f2bf(p1[r]);
        }
        asm volatile("s_wait_dscnt 0" ::: "memory");
        v16bf pa = cat16(ld8(&P[lmod * 32 + aoff0]), ld8(&P[lmod * 32 + aoff1]));

#pragma unroll
        for (int nt = 0; nt < 4; ++nt)
            o[nt] = wmma_bf16(pa, bv[nt], o[nt]);

        bk[0][0] = bkn[0][0]; bk[0][1] = bkn[0][1];
        bk[1][0] = bkn[1][0]; bk[1][1] = bkn[1][1];
    }
#pragma unroll
    for (int nt = 0; nt < 4; ++nt)
#pragma unroll
        for (int r = 0; r < 8; ++r) {
            int row = qt * 16 + r + lhalf * 8;
            oheads[((size_t)(b * TT) + row) * DD + h * 64 + nt * 16 + lmod] =
                f2bf(o[nt][r] / lsum[r]);
        }
}

// ------------------------------- launcher ----------------------------------
static void* carve(char*& p, size_t bytes) {
    void* r = p; p += (bytes + 255) & ~(size_t)255; return r;
}

extern "C" void kernel_launch(void* const* d_in, const int* in_sizes, int n_in,
                              void* d_out, int out_size, void* d_ws, size_t ws_size,
                              hipStream_t stream) {
    const int*   tokens = (const int*)d_in[0];
    const float* tokE   = (const float*)d_in[1];
    const float* posE   = (const float*)d_in[2];
    const float* ln1g   = (const float*)d_in[3];
    const float* ln1b   = (const float*)d_in[4];
    const float* Wq     = (const float*)d_in[5];
    const float* Wk     = (const float*)d_in[6];
    const float* Wv     = (const float*)d_in[7];
    const float* Wo     = (const float*)d_in[8];
    const float* ln2g   = (const float*)d_in[9];
    const float* ln2b   = (const float*)d_in[10];
    const float* W1     = (const float*)d_in[11];
    const float* b1     = (const float*)d_in[12];
    const float* W2     = (const float*)d_in[13];
    const float* b2     = (const float*)d_in[14];
    const float* lnfg   = (const float*)d_in[15];
    const float* lnfb   = (const float*)d_in[16];
    const float* Wlm    = (const float*)d_in[17];
    const float* blm    = (const float*)d_in[18];
    float* out = (float*)d_out;

    char* p = (char*)d_ws;
    float*  x      = (float*) carve(p, (size_t)MM * DD * 4);
    __bf16* hbuf   = (__bf16*)carve(p, (size_t)MM * DD * 2);
    __bf16* qkv    = (__bf16*)carve(p, (size_t)MM * QKVN * 2);
    __bf16* vpack  = (__bf16*)carve(p, (size_t)BB * HH * TT * DHD * 2);
    __bf16* oheads = (__bf16*)carve(p, (size_t)MM * DD * 2);
    __bf16* abuf   = (__bf16*)carve(p, (size_t)MM * FFD * 2);
    __bf16* qkvP   = (__bf16*)carve(p, (size_t)LL * QKVN * DD * 2);
    __bf16* woP    = (__bf16*)carve(p, (size_t)LL * DD * DD * 2);
    __bf16* w1P    = (__bf16*)carve(p, (size_t)LL * DD * FFD * 2);
    __bf16* w2P    = (__bf16*)carve(p, (size_t)LL * FFD * DD * 2);
    __bf16* wlmP   = (__bf16*)carve(p, (size_t)DD * VV * 2);

    // ---- pack all weights to bf16 B-tiles ----
    const int pThr = 256;
    for (int l = 0; l < LL; ++l) {
        int nQ = (DD / 32) * (DD / 16) * 32; // 36864
        hipLaunchKernelGGL(k_pack_b_qkv, dim3((nQ + pThr - 1) / pThr), dim3(pThr), 0, stream,
                           Wq + (size_t)l * HH * DD * DHD, qkvP + (size_t)l * QKVN * DD, 0);
        hipLaunchKernelGGL(k_pack_b_qkv, dim3((nQ + pThr - 1) / pThr), dim3(pThr), 0, stream,
                           Wk + (size_t)l * HH * DD * DHD, qkvP + (size_t)l * QKVN * DD, 48);
        hipLaunchKernelGGL(k_pack_b_qkv, dim3((nQ + pThr - 1) / pThr), dim3(pThr), 0, stream,
                           Wv + (size_t)l * HH * DD * DHD, qkvP + (size_t)l * QKVN * DD, 96);
        int nWo = (DD / 32) * (DD / 16) * 32;
        hipLaunchKernelGGL(k_pack_b_f32, dim3((nWo + pThr - 1) / pThr), dim3(pThr), 0, stream,
                           Wo + (size_t)l * DD * DD, woP + (size_t)l * DD * DD, DD, DD);
        int nW1 = (DD / 32) * (FFD / 16) * 32;
        hipLaunchKernelGGL(k_pack_b_f32, dim3((nW1 + pThr - 1) / pThr), dim3(pThr), 0, stream,
                           W1 + (size_t)l * DD * FFD, w1P + (size_t)l * DD * FFD, DD, FFD);
        int nW2 = (FFD / 32) * (DD / 16) * 32;
        hipLaunchKernelGGL(k_pack_b_f32, dim3((nW2 + pThr - 1) / pThr), dim3(pThr), 0, stream,
                           W2 + (size_t)l * FFD * DD, w2P + (size_t)l * FFD * DD, FFD, DD);
    }
    {
        int nLM = (DD / 32) * (VV / 16) * 32; // 1,536,000
        hipLaunchKernelGGL(k_pack_b_f32, dim3((nLM + pThr - 1) / pThr), dim3(pThr), 0, stream,
                           Wlm, wlmP, DD, VV);
    }

    // ---- embedding ----
    hipLaunchKernelGGL(k_embed, dim3(((size_t)MM * DD + 255) / 256), dim3(256), 0, stream,
                       tokens, tokE, posE, x);

    // ---- transformer layers ----
    for (int l = 0; l < LL; ++l) {
        hipLaunchKernelGGL(k_layernorm, dim3(MM), dim3(256), 0, stream,
                           x, ln1g + l * DD, ln1b + l * DD, hbuf);
        // QKV projection: [2048,768] x [768,2304] -> bf16 qkv
        hipLaunchKernelGGL(k_gemm, dim3(QKVN / 64, MM / 256), dim3(256), 0, stream,
                           hbuf, qkvP + (size_t)l * QKVN * DD,
                           (const float*)nullptr, (const float*)nullptr,
                           (float*)nullptr, qkv, QKVN, DD, 0);
        // pack V per head for P@V
        {
            int nV = BB * HH * (TT / 32) * 4 * 32;
            hipLaunchKernelGGL(k_pack_v, dim3((nV + pThr - 1) / pThr), dim3(pThr), 0, stream,
                               qkv, vpack);
        }
        // flash attention -> oheads bf16 [M, D] (col = h*64+e)
        hipLaunchKernelGGL(k_flash, dim3(TT / 16, (BB * HH) / 8), dim3(256), 0, stream,
                           qkv, vpack, oheads);
        // x = x + oheads @ Wo
        hipLaunchKernelGGL(k_gemm, dim3(DD / 64, MM / 256), dim3(256), 0, stream,
                           oheads, woP + (size_t)l * DD * DD,
                           (const float*)nullptr, x, x, (__bf16*)nullptr, DD, DD, 0);
        hipLaunchKernelGGL(k_layernorm, dim3(MM), dim3(256), 0, stream,
                           x, ln2g + l * DD, ln2b + l * DD, hbuf);
        // a = relu(h @ W1 + b1) -> bf16
        hipLaunchKernelGGL(k_gemm, dim3(FFD / 64, MM / 256), dim3(256), 0, stream,
                           hbuf, w1P + (size_t)l * DD * FFD,
                           b1 + l * FFD, (const float*)nullptr,
                           (float*)nullptr, abuf, FFD, DD, 1);
        // x = x + a @ W2 + b2
        hipLaunchKernelGGL(k_gemm, dim3(DD / 64, MM / 256), dim3(256), 0, stream,
                           abuf, w2P + (size_t)l * FFD * DD,
                           b2 + l * DD, x, x, (__bf16*)nullptr, DD, FFD, 0);
    }

    // ---- final LN + LM head ----
    hipLaunchKernelGGL(k_layernorm, dim3(MM), dim3(256), 0, stream, x, lnfg, lnfb, hbuf);
    hipLaunchKernelGGL(k_gemm, dim3(VV / 64, MM / 256), dim3(256), 0, stream,
                       hbuf, wlmP, blm, (const float*)nullptr,
                       out, (__bf16*)nullptr, VV, DD, 0);
}
